// AFE_78958678770209
// MI455X (gfx1250) — matrix operand order repaired and checked by reading, not compiled, not run on hardware
//
#include <hip/hip_runtime.h>

// Problem constants
#define BATCH 32
#define CIN   192
#define COUT  384
#define HH    56
#define WWW   56
#define S1    3136      // 56*56
#define M1    100352    // 32*3136
#define H2    28
#define S2    784       // 28*28
#define M2    25088     // 32*784
#define K3    768       // 192*4

typedef __attribute__((ext_vector_type(16))) __bf16 v16bf;
typedef __attribute__((ext_vector_type(8)))  __bf16 v8bf;
typedef __attribute__((ext_vector_type(8)))  float  v8f;

__device__ __forceinline__ __bf16 f2bf(float f) {
    unsigned u = __builtin_bit_cast(unsigned, f);
    u += 0x7FFFu + ((u >> 16) & 1u);              // round-to-nearest-even
    unsigned short s = (unsigned short)(u >> 16);
    return __builtin_bit_cast(__bf16, s);
}
__device__ __forceinline__ float bf2f(__bf16 h) {
    unsigned short s = __builtin_bit_cast(unsigned short, h);
    unsigned u = ((unsigned)s) << 16;
    return __builtin_bit_cast(float, u);
}

// ---------------- weight-fusion prep kernels (tiny). All weights stored [N][K] bf16.

// W12nk[o][c] = sum_m pw1_w[o,m] * dw1_w[m,c]
__global__ void prep_w12(const float* __restrict__ pw1_w,
                         const float* __restrict__ dw1_w,
                         __bf16* __restrict__ W12) {
    int c = blockIdx.x, o = threadIdx.x;
    float acc = 0.f;
    for (int m = 0; m < CIN; ++m)
        acc += pw1_w[o * CIN + m] * dw1_w[m * CIN + c];
    W12[o * CIN + c] = f2bf(acc);
}

// Wd[o][c] = def_w[o,c]  (straight bf16 convert, already [N][K])
__global__ void prep_wd(const float* __restrict__ def_w, __bf16* __restrict__ Wd) {
    int i = blockIdx.x * 256 + threadIdx.x;     // 36864 total
    Wd[i] = f2bf(def_w[i]);
}

// W34nk[o2][k = q*192 + i] = sum_o pw2_w[o2,o] * dw2_w[o,i,kh,kw],  q = kh*2+kw
__global__ void prep_w34(const float* __restrict__ pw2_w,
                         const float* __restrict__ dw2_w,
                         __bf16* __restrict__ W34) {
    int k = blockIdx.x, o2 = threadIdx.x;
    int q = k / CIN, i = k - q * CIN;
    int kh = q >> 1, kw = q & 1;
    float acc = 0.f;
    for (int o = 0; o < CIN; ++o)
        acc += pw2_w[o2 * CIN + o] * dw2_w[((o * CIN + i) * 2 + kh) * 2 + kw];
    W34[o2 * K3 + k] = f2bf(acc);
}

// fused biases
__global__ void prep_bias(const float* __restrict__ pw1_w, const float* __restrict__ dw1_b,
                          const float* __restrict__ pw1_b, const float* __restrict__ def_b,
                          const float* __restrict__ pw2_w, const float* __restrict__ dw2_b,
                          const float* __restrict__ pw2_b,
                          float* __restrict__ b12, float* __restrict__ bd,
                          float* __restrict__ b34) {
    int i = threadIdx.x;
    if (i < CIN) {
        float a = 0.f;
        for (int m = 0; m < CIN; ++m) a += pw1_w[i * CIN + m] * dw1_b[m];
        b12[i] = a + pw1_b[i];
        bd[i]  = def_b[i];
    }
    float a = 0.f;
    for (int o = 0; o < CIN; ++o) a += pw2_w[i * CIN + o] * dw2_b[o];
    b34[i] = a + pw2_b[i];
}

// ---------------- offsets: 3x3 conv x -> [B,2,56,56] (x is L2-resident: 77MB < 192MB) ----
__global__ __launch_bounds__(256) void offsets_kernel(
        const float* __restrict__ x, const float* __restrict__ pw,
        const float* __restrict__ pb, float* __restrict__ off) {
    int gid = blockIdx.x * 256 + threadIdx.x;            // 200704 total
    int b  = gid / (2 * S1);
    int oc = (gid / S1) & 1;
    int s  = gid % S1;
    int h = s / WWW, w = s % WWW;
    float acc = pb[oc];
    for (int c = 0; c < CIN; ++c) {
        const float* xp = x + ((size_t)(b * CIN + c)) * S1;
        const float* wp = pw + (oc * CIN + c) * 9;
#pragma unroll
        for (int kh = 0; kh < 3; ++kh) {
            int ih = h + kh - 1;
            if (ih < 0 || ih >= HH) continue;
#pragma unroll
            for (int kw = 0; kw < 3; ++kw) {
                int iw = w + kw - 1;
                if (iw < 0 || iw >= WWW) continue;
                acc += wp[kh * 3 + kw] * xp[ih * WWW + iw];
            }
        }
    }
    off[((size_t)b * 2 + oc) * S1 + s] = acc;
}

// ---------------- deformable bilinear gather (NHWC bf16 -> NHWC bf16) ----------------
__global__ __launch_bounds__(256) void gather_kernel(
        const __bf16* __restrict__ y1, const float* __restrict__ off,
        __bf16* __restrict__ y2) {
    int gid = blockIdx.x * 256 + threadIdx.x;            // M1*24
    int m = gid / 24;
    int g = gid - m * 24;                                 // 8-channel group
    int b = m / S1, s = m - b * S1;
    int h = s / WWW, w = s % WWW;
    float dy = off[((size_t)b * 2 + 0) * S1 + s];
    float dx = off[((size_t)b * 2 + 1) * S1 + s];
    float py = fminf(fmaxf((float)h + dy, 0.f), (float)(HH - 1));
    float px = fminf(fmaxf((float)w + dx, 0.f), (float)(WWW - 1));
    float y0f = floorf(py), x0f = floorf(px);
    int y0 = (int)y0f, x0 = (int)x0f;
    int y1i = min(y0 + 1, HH - 1), x1i = min(x0 + 1, WWW - 1);
    float wy = py - y0f, wx = px - x0f;
    float w00 = (1.f - wy) * (1.f - wx), w01 = (1.f - wy) * wx;
    float w10 = wy * (1.f - wx),         w11 = wy * wx;
    size_t base = (size_t)b * S1 * CIN;
    const v8bf* p00 = (const v8bf*)(y1 + base + ((size_t)y0  * WWW + x0 ) * CIN + g * 8);
    const v8bf* p01 = (const v8bf*)(y1 + base + ((size_t)y0  * WWW + x1i) * CIN + g * 8);
    const v8bf* p10 = (const v8bf*)(y1 + base + ((size_t)y1i * WWW + x0 ) * CIN + g * 8);
    const v8bf* p11 = (const v8bf*)(y1 + base + ((size_t)y1i * WWW + x1i) * CIN + g * 8);
    v8bf a = *p00, bb = *p01, c = *p10, d = *p11, r;
#pragma unroll
    for (int j = 0; j < 8; ++j) {
        float v = w00 * bf2f(a[j]) + w01 * bf2f(bb[j]) + w10 * bf2f(c[j]) + w11 * bf2f(d[j]);
        r[j] = f2bf(v);
    }
    *(v8bf*)(y2 + (size_t)m * CIN + g * 8) = r;
}

// ---------------- WMMA bf16 GEMM: C[M,N] = A[M,K] * W[N,K]^T + bias ----------------
// LDS tiles are stored in WMMA *fragment order*:
//   A 16x32 bf16 tile, lane = m%16 + 16*((k>>3)&1), element e = (k&7) + 8*(k>>4)
//     -> each m-row's four k-octets are contiguous 8-element runs: 4x ds_store_b128 to stage,
//        each lane reads its whole v16bf fragment as one 32B LDS load (2x ds_load_b128).
//   B 32x16 bf16 tile (same k order as A), lane = n%16 + 16*((k>>3)&1), element e = (k&7) + 8*(k>>4).
// MODE 0: A = x (f32, NCHW strided), out bf16 NHWC
// MODE 1: A = bf16 row-major [M,192], out bf16 NHWC
// MODE 2: A = y3 bf16 NHWC 2x2 patches (K=768), out f32 NCHW [B,384,28,28]
#define BM 256
#define BN 64
#define BK 32

template <int MODE>
__global__ __launch_bounds__(256) void gemm_bf16(
        const float* __restrict__ Af32, const __bf16* __restrict__ Abf,
        const __bf16* __restrict__ Wnk, const float* __restrict__ bias,
        __bf16* __restrict__ Obf, float* __restrict__ Of32,
        int Ktot, int Ntot) {
    const int t    = threadIdx.x;
    const int lane = t & 31;
    const int wv   = t >> 5;
    const int half = lane >> 4;
    const int l16  = lane & 15;
    const int m0   = blockIdx.x * BM;
    const int n0   = blockIdx.y * BN;
    const int wvM  = (wv & 3) * 64;          // 4 waves along M
    const int wvN  = (wv >> 2) * 32;         // 2 waves along N

    // fragment-order LDS: [tile][lane][element]
    __shared__ __attribute__((aligned(32))) __bf16 Asf[16][32][16];  // 16 KB
    __shared__ __attribute__((aligned(32))) __bf16 Bsf[4][32][16];   //  4 KB

    v8f acc[4][2] = {};

    // per-thread A row m = m0 + t
    const int mA = m0 + t;
    int bA = 0, sA = 0;
    int pb0 = 0, pb1 = 0, pb2 = 0, pb3 = 0;
    if (MODE == 0) { bA = mA / S1; sA = mA - bA * S1; }
    if (MODE == 2) {
        int b = mA / S2, r = mA - b * S2;
        int oh = r / H2, ow = r - oh * H2;
        pb0 = ((b * S1) + (2 * oh) * WWW + 2 * ow) * CIN;   // (kh=0,kw=0)
        pb1 = pb0 + CIN;                                    // (0,1)
        pb2 = pb0 + WWW * CIN;                              // (1,0)
        pb3 = pb2 + CIN;                                    // (1,1)
    }
    const int amt = t >> 4, al = t & 15;      // A staging: m-tile, lane-low
    const int bnl = t >> 2, bkc = (t & 3) * 8;// B staging: n-local 0..63, k-chunk {0,8,16,24}

    for (int k0 = 0; k0 < Ktot; k0 += BK) {
        // ---- stage A tile: 4 x b128 stores per thread in fragment order ----
        v8bf ch[4];                            // k octets 0-7, 8-15, 16-23, 24-31
        if (MODE == 0) {
            const float* src = Af32 + ((size_t)(bA * CIN + k0)) * S1 + sA;
            if (k0 + BK < Ktot) __builtin_prefetch(src + (size_t)BK * S1, 0, 0);
#pragma unroll
            for (int c = 0; c < 4; ++c)
#pragma unroll
                for (int j = 0; j < 8; ++j)
                    ch[c][j] = f2bf(src[(size_t)(c * 8 + j) * S1]);
        } else {
            const __bf16* src;
            if (MODE == 1) {
                src = Abf + (size_t)mA * CIN + k0;
            } else {
                int q  = k0 / CIN;             // constant over the 32 (192 % 32 == 0)
                int c0 = k0 - q * CIN;
                src = Abf + (q == 0 ? pb0 : q == 1 ? pb1 : q == 2 ? pb2 : pb3) + c0;
            }
#pragma unroll
            for (int c = 0; c < 4; ++c) ch[c] = *(const v8bf*)(src + c * 8);
        }
        *(v8bf*)&Asf[amt][al     ][0] = ch[0];   // k 0..7   (half 0, e 0..7)
        *(v8bf*)&Asf[amt][al + 16][0] = ch[1];   // k 8..15  (half 1, e 0..7)
        *(v8bf*)&Asf[amt][al     ][8] = ch[2];   // k 16..23 (half 0, e 8..15)
        *(v8bf*)&Asf[amt][al + 16][8] = ch[3];   // k 24..31 (half 1, e 8..15)

        // ---- stage B tile: 1 x b128 per thread (weights are [N][K], contiguous k) ----
        {
            v8bf wv8 = *(const v8bf*)(Wnk + (size_t)(n0 + bnl) * Ktot + k0 + bkc);
            *(v8bf*)&Bsf[bnl >> 4][(bnl & 15) + 16 * ((bkc >> 3) & 1)][8 * (bkc >> 4)] = wv8;
        }
        __syncthreads();

        // ---- fragments: one 32B LDS read each ----
        v16bf afr[4], bfr[2];
#pragma unroll
        for (int mt = 0; mt < 4; ++mt)
            afr[mt] = *(const v16bf*)&Asf[(wvM >> 4) + mt][lane][0];
#pragma unroll
        for (int nt = 0; nt < 2; ++nt)
            bfr[nt] = *(const v16bf*)&Bsf[(wvN >> 4) + nt][lane][0];

#pragma unroll
        for (int mt = 0; mt < 4; ++mt)
#pragma unroll
            for (int nt = 0; nt < 2; ++nt)
                acc[mt][nt] = __builtin_amdgcn_wmma_f32_16x16x32_bf16(
                    false, afr[mt], false, bfr[nt], (short)0, acc[mt][nt], false, false);
        __syncthreads();
    }

    // ---- epilogue ----
#pragma unroll
    for (int nt = 0; nt < 2; ++nt) {
        int n = n0 + wvN + nt * 16 + l16;
        float bv = bias[n];
#pragma unroll
        for (int mt = 0; mt < 4; ++mt) {
#pragma unroll
            for (int r = 0; r < 8; ++r) {
                int m = m0 + wvM + mt * 16 + r + half * 8;
                float v = acc[mt][nt][r] + bv;
                if (MODE == 2) {
                    int b = m / S2, rr = m - b * S2;
                    Of32[((size_t)b * COUT + n) * S2 + rr] = v;   // NCHW [32,384,28,28]
                } else {
                    Obf[(size_t)m * CIN + n] = f2bf(v);           // NHWC bf16
                }
            }
        }
    }
}

// ---------------- launch ----------------
extern "C" void kernel_launch(void* const* d_in, const int* in_sizes, int n_in,
                              void* d_out, int out_size, void* d_ws, size_t ws_size,
                              hipStream_t stream) {
    const float* x      = (const float*)d_in[0];
    const float* dw1_w  = (const float*)d_in[1];
    const float* dw1_b  = (const float*)d_in[2];
    const float* pw1_w  = (const float*)d_in[3];
    const float* pw1_b  = (const float*)d_in[4];
    const float* poff_w = (const float*)d_in[5];
    const float* poff_b = (const float*)d_in[6];
    const float* def_w  = (const float*)d_in[7];
    const float* def_b  = (const float*)d_in[8];
    const float* dw2_w  = (const float*)d_in[9];
    const float* dw2_b  = (const float*)d_in[10];
    const float* pw2_w  = (const float*)d_in[11];
    const float* pw2_b  = (const float*)d_in[12];

    char* ws = (char*)d_ws;
    __bf16* W12 = (__bf16*)(ws + 0);           //  73,728 B  [192][192]
    __bf16* Wd  = (__bf16*)(ws + 73728);       //  73,728 B  [192][192]
    __bf16* W34 = (__bf16*)(ws + 147456);      // 589,824 B  [384][768]
    float*  b12 = (float*)(ws + 737280);
    float*  bd  = (float*)(ws + 738048);
    float*  b34 = (float*)(ws + 738816);
    float*  off = (float*)(ws + 740352);       // 802,816 B
    __bf16* y1  = (__bf16*)(ws + 1543168);     // 38.5 MB NHWC bf16
    __bf16* y2  = (__bf16*)(ws + 40078336);    // 38.5 MB
    __bf16* y3  = y1;                          // reuse (y1 dead after gather)

    prep_w12 <<<192, 192, 0, stream>>>(pw1_w, dw1_w, W12);
    prep_wd  <<<144, 256, 0, stream>>>(def_w, Wd);           // 36864 elements
    prep_w34 <<<K3,  COUT, 0, stream>>>(pw2_w, dw2_w, W34);
    prep_bias<<<1,   COUT, 0, stream>>>(pw1_w, dw1_b, pw1_b, def_b,
                                        pw2_w, dw2_b, pw2_b, b12, bd, b34);
    offsets_kernel<<<784, 256, 0, stream>>>(x, poff_w, poff_b, off);

    // fused dw1+pw1: y1 = x * W12^T + b12        (M=100352, K=192, N=192)
    gemm_bf16<0><<<dim3(M1 / BM, CIN / BN), 256, 0, stream>>>(
        x, nullptr, W12, b12, y1, nullptr, CIN, CIN);
    // deformable bilinear gather
    gather_kernel<<<(M1 * 24) / 256, 256, 0, stream>>>(y1, off, y2);
    // def 1x1: y3 = y2 * Wd^T + bd               (M=100352, K=192, N=192)
    gemm_bf16<1><<<dim3(M1 / BM, CIN / BN), 256, 0, stream>>>(
        nullptr, y2, Wd, bd, y3, nullptr, CIN, CIN);
    // fused dw2+pw2: out = patches(y3) * W34^T + b34  (M=25088, K=768, N=384)
    gemm_bf16<2><<<dim3(M2 / BM, COUT / BN), 256, 0, stream>>>(
        nullptr, y3, W34, b34, nullptr, (float*)d_out, K3, COUT);
}
